// PointTransformerModule_1305670058591
// MI455X (gfx1250) — matrix operand order repaired and checked by reading, not compiled, not run on hardware
//
#include <hip/hip_runtime.h>
#include <hip/hip_bf16.h>
#include <math.h>

// ---------------------------------------------------------------------------
// PointTransformer module for MI455X (gfx1250, wave32, WMMA).
//   B=8, N=4096, S=2048, K=10, C=64, R=0.25
// Kernels:
//   zero_kernel   : clear d_out (rows >= B*S stay zero, faithful to scatter-init)
//   fps_kernel    : farthest point sampling, 1 block/batch, point cloud in LDS
//   radius_kernel : K-nearest within radius, LDS point tiles + global prefetch
//   attn_kernel   : fused projections + logit GEMM (v_wmma_f32_16x16x32_f16)
//                   + masked softmax over K + aggregate, direct store (unique tgt)
// ---------------------------------------------------------------------------

#define BB   8
#define NN   4096
#define SS   2048
#define KK   10
#define CC   64
#define R2f  0.0625f

typedef __attribute__((ext_vector_type(16))) _Float16 v16h;
typedef __attribute__((ext_vector_type(8)))  _Float16 v8h;
typedef __attribute__((ext_vector_type(8)))  float    v8f;

// --------------------------------------------------------------------------
__global__ void zero_kernel(float* __restrict__ out, int n) {
    int i = blockIdx.x * blockDim.x + threadIdx.x;
    if (i < n) out[i] = 0.f;
}

// --------------------------------------------------------------------------
// FPS: one block per batch, 512 threads (16 wave32), points resident in LDS.
__global__ __launch_bounds__(512)
void fps_kernel(const float* __restrict__ pos, int* __restrict__ idxOut) {
    constexpr int NT = 512, PPT = NN / NT;         // 8 points per thread
    __shared__ float sx[NN], sy[NN], sz[NN];       // 48 KB
    __shared__ float rv[16];
    __shared__ int   ri[16];
    __shared__ int   sLast;

    const int b   = blockIdx.x;
    const int tid = threadIdx.x;
    const float* p = pos + (size_t)b * NN * 3;

    for (int i = tid; i < NN; i += NT) {
        sx[i] = p[i * 3 + 0]; sy[i] = p[i * 3 + 1]; sz[i] = p[i * 3 + 2];
    }
    float mind[PPT];
#pragma unroll
    for (int u = 0; u < PPT; ++u) mind[u] = 3.4e38f;
    if (tid == 0) { idxOut[b * SS] = 0; sLast = 0; }
    __syncthreads();

    int last = 0;
    for (int s = 1; s < SS; ++s) {
        const float lx = sx[last], ly = sy[last], lz = sz[last];
        float best = -1.f; int bi = 0x7fffffff;
#pragma unroll
        for (int u = 0; u < PPT; ++u) {
            const int i = tid + u * NT;
            float dx = sx[i] - lx, dy = sy[i] - ly, dz = sz[i] - lz;
            float d  = fmaf(dz, dz, fmaf(dy, dy, dx * dx));
            float mv = fminf(mind[u], d);
            mind[u]  = mv;
            if (mv > best) { best = mv; bi = i; }
        }
        // wave32 butterfly argmax (tie -> lowest index, matching jnp.argmax)
#pragma unroll
        for (int off = 16; off > 0; off >>= 1) {
            float ov = __shfl_xor(best, off, 32);
            int   oi = __shfl_xor(bi,   off, 32);
            if (ov > best || (ov == best && oi < bi)) { best = ov; bi = oi; }
        }
        const int wv = tid >> 5;
        if ((tid & 31) == 0) { rv[wv] = best; ri[wv] = bi; }
        __syncthreads();
        if (tid < 32) {
            float v2 = (tid < 16) ? rv[tid] : -1.f;
            int   i2 = (tid < 16) ? ri[tid] : 0x7fffffff;
#pragma unroll
            for (int off = 8; off > 0; off >>= 1) {
                float ov = __shfl_xor(v2, off, 32);
                int   oi = __shfl_xor(i2, off, 32);
                if (ov > v2 || (ov == v2 && oi < i2)) { v2 = ov; i2 = oi; }
            }
            if (tid == 0) { sLast = i2; idxOut[b * SS + s] = i2; }
        }
        __syncthreads();
        last = sLast;
    }
}

// --------------------------------------------------------------------------
// Radius / K-nearest: one thread per query, point tiles staged through LDS.
__global__ __launch_bounds__(256)
void radius_kernel(const float* __restrict__ pos, const int* __restrict__ idxIn,
                   int* __restrict__ nbrOut) {
    constexpr int TILE = 1024;
    __shared__ float tx[TILE], ty[TILE], tz[TILE];   // 12 KB

    const int q = blockIdx.x * 256 + threadIdx.x;    // 0 .. B*S-1
    const int b = q >> 11;                           // / SS (blocks don't straddle b)
    const float* p = pos + (size_t)b * NN * 3;
    const int qi = idxIn[q];
    const float qx = p[qi * 3 + 0], qy = p[qi * 3 + 1], qz = p[qi * 3 + 2];

    float dk[KK]; int ik[KK];
#pragma unroll
    for (int u = 0; u < KK; ++u) { dk[u] = 3.4e38f; ik[u] = -1; }

    for (int t0 = 0; t0 < NN; t0 += TILE) {
        __syncthreads();
        for (int i = threadIdx.x; i < TILE; i += 256) {
            tx[i] = p[(t0 + i) * 3 + 0];
            ty[i] = p[(t0 + i) * 3 + 1];
            tz[i] = p[(t0 + i) * 3 + 2];
        }
        if (t0 + TILE < NN)
            __builtin_prefetch(p + (size_t)(t0 + TILE) * 3, 0, 1);  // global_prefetch_b8
        __syncthreads();
        for (int j = 0; j < TILE; ++j) {
            float dx = tx[j] - qx, dy = ty[j] - qy, dz = tz[j] - qz;
            float d  = fmaf(dz, dz, fmaf(dy, dy, dx * dx));
            if (d <= R2f && d < dk[KK - 1]) {         // keep K nearest in radius
                dk[KK - 1] = d; ik[KK - 1] = t0 + j;
#pragma unroll
                for (int u = KK - 1; u > 0; --u) {    // bubble into sorted list
                    if (dk[u] < dk[u - 1]) {
                        float td = dk[u]; dk[u] = dk[u - 1]; dk[u - 1] = td;
                        int   ti = ik[u]; ik[u] = ik[u - 1]; ik[u - 1] = ti;
                    }
                }
            }
        }
    }
#pragma unroll
    for (int u = 0; u < KK; ++u) nbrOut[q * KK + u] = ik[u];   // -1 == masked
}

// --------------------------------------------------------------------------
// Fused attention: 8 targets (80 edge rows) per block of 128 threads (4 waves).
#define GT    8
#define ROWS  (GT * KK)    // 80
#define RT    5            // 80 / 16 row-tiles
#define ASTR  72           // sA half-stride (144B = 9*16B -> 16B-aligned chunks)

__global__ __launch_bounds__(128)
void attn_kernel(const float* __restrict__ pos,
                 const float* __restrict__ w_pos,  const float* __restrict__ b_pos,
                 const float* __restrict__ w_attn, const float* __restrict__ b_attn,
                 const float* __restrict__ w_lin,  const float* __restrict__ w_src,
                 const float* __restrict__ w_dst,
                 const int*   __restrict__ nbr,    float* __restrict__ out) {
    __shared__ _Float16 sWa[CC * CC];                         // w_attn, f16, 8 KB
    __shared__ float sWpos[3 * CC], sWsrc[3 * CC], sWlin[3 * CC], sWdst[3 * CC];
    __shared__ float sBpos[CC], sBattn[CC];
    __shared__ __align__(16) _Float16 sA[ROWS * ASTR];        // A panel f16, 11.25 KB
    __shared__ float    sW[ROWS * CC];                        // xv[src]+delta, 20 KB
    __shared__ _Float16 sL[ROWS * CC];                        // logits f16, 10 KB
    __shared__ int      sNbr[ROWS];

    const int tid  = threadIdx.x;
    const int lane = tid & 31;
    const int wv   = tid >> 5;            // wave id == column tile nb (0..3)
    const int kh   = lane >> 4;           // half-wave group
    const int ncol = (wv << 4) + (lane & 15);
    const int g0   = blockIdx.x * GT;     // first target (== flat tgt row b*S+s)

    for (int i = tid; i < CC * CC; i += 128) sWa[i] = (_Float16)w_attn[i];
    for (int i = tid; i < 3 * CC; i += 128) {
        sWpos[i] = w_pos[i]; sWsrc[i] = w_src[i];
        sWlin[i] = w_lin[i]; sWdst[i] = w_dst[i];
    }
    if (tid < CC) { sBpos[tid] = b_pos[tid]; sBattn[tid] = b_attn[tid]; }
    if (tid < ROWS) sNbr[tid] = nbr[g0 * KK + tid];
    __syncthreads();

    // ---- B fragments of w_attn (32x16 f16, K halves contiguous per lane group),
    //      built once and pinned in registers for the whole kernel.
    v16h bf0, bf1;
    {
        union { v16h v; _Float16 h[16]; } u0, u1;
#pragma unroll
        for (int e = 0; e < 16; ++e) {
            u0.h[e] = sWa[(kh * 16 + e) * CC + ncol];          // K = 0..31
            u1.h[e] = sWa[(32 + kh * 16 + e) * CC + ncol];     // K = 32..63
        }
        bf0 = u0.v; bf1 = u1.v;
    }

    // ---- Phase A: build A panel (f16) and xv+delta panel (f32) on the fly.
    {
        const int c  = tid & 63;
        const int rh = tid >> 6;
        for (int r = rh; r < ROWS; r += 2) {
            const int t  = r / KK, k = r - t * KK;
            const int tg = g0 + t;                 // flat tgt row (b*S+s), faithful
            const int b  = tg >> 11;
            const int j  = sNbr[t * KK + k];
            float av = 0.f, wvv = 0.f;
            if (j >= 0) {
                const float pix = pos[tg * 3 + 0], piy = pos[tg * 3 + 1], piz = pos[tg * 3 + 2];
                const int   src = b * NN + j;
                const float pjx = pos[src * 3 + 0], pjy = pos[src * 3 + 1], pjz = pos[src * 3 + 2];
                const float dx = pix - pjx, dy = piy - pjy, dz = piz - pjz;
                float delta = sBpos[c] + dx * sWpos[c] + dy * sWpos[CC + c] + dz * sWpos[2 * CC + c];
                float asrc  = pjx * sWsrc[c] + pjy * sWsrc[CC + c] + pjz * sWsrc[2 * CC + c];
                float adst  = pix * sWdst[c] + piy * sWdst[CC + c] + piz * sWdst[2 * CC + c];
                float xv    = pjx * sWlin[c] + pjy * sWlin[CC + c] + pjz * sWlin[2 * CC + c];
                av  = adst - asrc + delta;
                wvv = xv + delta;
            }
            sA[r * ASTR + c] = (_Float16)av;
            sW[r * CC + c]   = wvv;
        }
    }
    __syncthreads();

    // ---- Phase B: logits = A @ w_attn + b_attn via v_wmma_f32_16x16x32_f16.
    {
        const float bb = sBattn[ncol];
#pragma unroll
        for (int rt = 0; rt < RT; ++rt) {
            const int row = rt * 16 + (lane & 15);
            // f16 A 16x32, contiguous k order chosen here (same as bf0/bf1): lanes 0-15 hold K=0..15,
            //                     lanes 16-31 hold K=16..31.
            union { v16h v; v8h p[2]; } a0, a1;
            a0.p[0] = *(const v8h*)(sA + row * ASTR + kh * 16);
            a0.p[1] = *(const v8h*)(sA + row * ASTR + 8 + kh * 16);
            a1.p[0] = *(const v8h*)(sA + row * ASTR + 32 + kh * 16);
            a1.p[1] = *(const v8h*)(sA + row * ASTR + 40 + kh * 16);
            v8f acc = { bb, bb, bb, bb, bb, bb, bb, bb };      // + b_attn[ncol]
            acc = __builtin_amdgcn_wmma_f32_16x16x32_f16(false, a0.v, false, bf0,
                                                         (short)0, acc, false, false);
            acc = __builtin_amdgcn_wmma_f32_16x16x32_f16(false, a1.v, false, bf1,
                                                         (short)0, acc, false, false);
            const int mb = rt * 16 + kh * 8;                   // D rows of this lane
#pragma unroll
            for (int v = 0; v < 8; ++v)
                sL[(mb + v) * CC + ncol] = (_Float16)acc[v];
        }
    }
    __syncthreads();

    // ---- Phase C: masked softmax over K per (target, channel) + aggregate.
    for (int it = tid; it < GT * CC; it += 128) {
        const int t = it >> 6, c = it & 63;
        float lv[KK]; bool va[KK];
        float m = -__builtin_inff();
#pragma unroll
        for (int k = 0; k < KK; ++k) {
            va[k] = sNbr[t * KK + k] >= 0;
            lv[k] = va[k] ? (float)sL[(t * KK + k) * CC + c] : -__builtin_inff();
            m = fmaxf(m, lv[k]);
        }
        if (!(m > -__builtin_inff())) m = 0.f;     // all masked -> m := 0
        float ssum = 0.f, acc = 0.f;
#pragma unroll
        for (int k = 0; k < KK; ++k) {
            float e = va[k] ? __expf(lv[k] - m) : 0.f;
            ssum += e;
            acc  += e * sW[(t * KK + k) * CC + c];
        }
        acc /= fmaxf(ssum, 1e-16f);
        out[(size_t)(g0 + t) * CC + c] = acc;      // tgt rows unique: direct store
    }
}

// --------------------------------------------------------------------------
extern "C" void kernel_launch(void* const* d_in, const int* in_sizes, int n_in,
                              void* d_out, int out_size, void* d_ws, size_t ws_size,
                              hipStream_t stream) {
    (void)in_sizes; (void)n_in; (void)out_size; (void)ws_size;
    const float* pos    = (const float*)d_in[0];
    const float* w_pos  = (const float*)d_in[1];
    const float* b_pos  = (const float*)d_in[2];
    const float* w_attn = (const float*)d_in[3];
    const float* b_attn = (const float*)d_in[4];
    const float* w_lin  = (const float*)d_in[5];
    const float* w_src  = (const float*)d_in[6];
    const float* w_dst  = (const float*)d_in[7];
    float* out = (float*)d_out;

    int* idxw = (int*)d_ws;              // [B*S]
    int* nbrw = idxw + BB * SS;          // [B*S*K]

    const int outN = BB * NN * CC;       // 2,097,152
    zero_kernel<<<(outN + 255) / 256, 256, 0, stream>>>(out, outN);
    fps_kernel<<<BB, 512, 0, stream>>>(pos, idxw);
    radius_kernel<<<(BB * SS) / 256, 256, 0, stream>>>(pos, idxw, nbrw);
    attn_kernel<<<(BB * SS) / GT, 128, 0, stream>>>(pos, w_pos, b_pos, w_attn, b_attn,
                                                    w_lin, w_src, w_dst, nbrw, out);
}
